// ConvolutionBlock_46325517254756
// MI455X (gfx1250) — matrix operand order repaired and checked by reading, not compiled, not run on hardware
//
#include <hip/hip_runtime.h>
#include <hip/hip_bf16.h>
#include <hip/hip_fp16.h>

// Problem constants (from reference): L=4, P=5, T=15, CIN=COUT=32, N=1000
#define NT_ 1000
#define TT_ 15
#define PP_ 5
#define CIN_ 32
#define COUT_ 32

// GEMM dims (per p): M=1000 (pad 1024), K=9600 = 2*10*15*32, N=640 = 2*10*32
#define KDIM 9600
#define NCOL 640
#define NGROUPS 20    // (ri, q2) groups; each spans 15 t-steps of K=32

typedef _Float16 half16 __attribute__((ext_vector_type(16)));
typedef _Float16 half8  __attribute__((ext_vector_type(8)));
typedef float    fv8    __attribute__((ext_vector_type(8)));
typedef float    fv4    __attribute__((ext_vector_type(4)));
typedef float    v8f    __attribute__((ext_vector_type(8)));

struct Group {
  const float* ptr;              // x[2*l2+ri] + k*P*CIN   (t=0 base)
  unsigned long long rowStride;  // elements per n: T*(l2+1)*P*CIN
  unsigned long long tStep;      // elements per t: (l2+1)*P*CIN
};

struct Params {
  const float* x[8];   // in{l}_{real,imag}  [N,T,l+1,P,CIN]
  const float* w[8];   // w{l}_{real,imag}   [l+1,T,P,CIN,COUT]
  const float* b[8];   // b{l}_{real,imag}   [1,l+1,P,COUT]
  float* out;          // concatenated tuple outputs
  float* k4;           // ws: 6400 f32 CG coefficient table
  Group* grp;          // ws: 20 K-group descriptors
  _Float16* weff;      // ws: [p][col][K] f16, 5*640*9600
};

__device__ __constant__ unsigned long long OUT_OFF[8] = {
  0ull, 160000ull, 320000ull, 640000ull, 960000ull, 1440000ull, 1920000ull, 2560000ull
};

// ---------------- CG coefficients (Racah formula, f64) ----------------
__device__ inline double dfact(int n) {
  double f = 1.0;
  for (int i = 2; i <= n; ++i) f *= (double)i;
  return f;
}

__device__ double cgcoef(int j1, int m1, int j2, int m2, int J, int M) {
  if (m1 + m2 != M) return 0.0;
  int lo = j1 - j2; if (lo < 0) lo = -lo;
  if (J < lo || J > j1 + j2) return 0.0;
  int am1 = m1 < 0 ? -m1 : m1, am2 = m2 < 0 ? -m2 : m2, aM = M < 0 ? -M : M;
  if (am1 > j1 || am2 > j2 || aM > J) return 0.0;
  double pre = (2.0 * J + 1.0) * dfact(J + j1 - j2) * dfact(J - j1 + j2)
             * dfact(j1 + j2 - J) / dfact(j1 + j2 + J + 1);
  pre *= dfact(J + M) * dfact(J - M) * dfact(j1 - m1) * dfact(j1 + m1)
       * dfact(j2 - m2) * dfact(j2 + m2);
  double s = 0.0;
  for (int k = 0; k <= j1 + j2 - J; ++k) {
    int d0 = k, d1 = j1 + j2 - J - k, d2 = j1 - m1 - k, d3 = j2 + m2 - k;
    int d4 = J - j2 + m1 + k, d5 = J - j1 - m2 + k;
    if (d0 < 0 || d1 < 0 || d2 < 0 || d3 < 0 || d4 < 0 || d5 < 0) continue;
    double den = dfact(d0) * dfact(d1) * dfact(d2) * dfact(d3) * dfact(d4) * dfact(d5);
    s += ((k & 1) ? -1.0 : 1.0) / den;
  }
  return sqrt(pre) * s;
}

__device__ inline int q_to_deg(int q) { return (q >= 6) ? 3 : ((q >= 3) ? 2 : ((q >= 1) ? 1 : 0)); }

// ---------------- Kernel 1: CG table K4 + K-group metadata ----------------
// K4 layout: idx = (((ri_o*2 + ri_x)*10 + q)*10 + q2)*16 + l1*4 + m
__global__ void setup_kernel(Params prm) {
  int idx = blockIdx.x * blockDim.x + threadIdx.x;
  if (idx < 6400) {
    const double PI_ = 3.14159265358979323846;
    int m    = idx & 3;
    int l1   = (idx >> 2) & 3;
    int q2   = (idx / 16) % 10;
    int q    = (idx / 160) % 10;
    int ri_x = (idx / 1600) & 1;
    int ri_o = (idx / 3200) & 1;
    int l  = q_to_deg(q);  int lm = q  - l  * (l  + 1) / 2;
    int l2 = q_to_deg(q2); int k  = q2 - l2 * (l2 + 1) / 2;
    int w_ri = ri_o ^ ri_x;
    float val = 0.0f;
    int dlo = l - l1; if (dlo < 0) dlo = -dlo;
    int dhi = l + l1; if (dhi > 3) dhi = 3;
    // imag-weight mask: w{l1}_imag has m==0 row zeroed
    if (m <= l1 && l2 >= dlo && l2 <= dhi && !(w_ri == 1 && m == 0)) {
      double c = 8.0 * PI_ * PI_ / (2.0 * l1 + 1.0)
               * sqrt((2.0 * l + 1.0) * (2.0 * l1 + 1.0) / (4.0 * PI_ * (2.0 * l2 + 1.0)))
               * cgcoef(l, 0, l1, 0, l2, 0);
      double t1 = 0.0, t2 = 0.0, t3 = 0.0;
      if (lm + m == k) t1 = cgcoef(l, lm, l1, m, l2, lm + m) * ((m & 1) ? -1.0 : 1.0);
      if (m > 0 && lm - m >= 0 && lm - m == k)
        t2 = cgcoef(l, lm, l1, -m, l2, lm - m) * ((l1 & 1) ? -1.0 : 1.0);
      if (m > 0 && lm - m < 0 && m - lm == k)
        t3 = (((m - lm) & 1) ? -1.0 : 1.0) * (((l1 + l2) & 1) ? -1.0 : 1.0)
           * cgcoef(l, lm, l1, -m, l2, lm - m);
      double v;
      if (ri_o == 0 && ri_x == 0)      v =  c * (t1 + t2 + t3);  // xr*wr -> out_r
      else if (ri_o == 0)              v = -c * (t1 - t2 + t3);  // xi*wi -> out_r
      else if (ri_x == 0)              v =  c * (t1 - t2 - t3);  // xr*wi -> out_i
      else                             v =  c * (t1 + t2 - t3);  // xi*wr -> out_i
      val = (float)v;
    }
    prm.k4[idx] = val;
  } else if (idx < 6400 + NGROUPS) {
    int g  = idx - 6400;              // 0..19
    int ri = g / 10;
    int q2 = g % 10;
    int l2 = q_to_deg(q2);
    int k  = q2 - l2 * (l2 + 1) / 2;
    Group gr;
    gr.ptr = prm.x[2 * l2 + ri] + (size_t)k * (PP_ * CIN_);
    gr.rowStride = (unsigned long long)(TT_ * (l2 + 1) * PP_ * CIN_);
    gr.tStep     = (unsigned long long)((l2 + 1) * PP_ * CIN_);
    prm.grp[g] = gr;
  }
}

// ---------------- Kernel 2: build effective weight matrix (f16) ----------------
// Weff[p][col][kidx], col = ri_o*320 + q*32 + j,  kidx = ri_x*4800 + (q2*15+t)*32 + i
__global__ void build_weff(Params prm) {
  long long idx = (long long)blockIdx.x * blockDim.x + threadIdx.x;
  if (idx >= (long long)PP_ * NCOL * KDIM) return;
  int kidx = (int)(idx % KDIM);
  int rest = (int)(idx / KDIM);
  int col  = rest % NCOL;
  int p    = rest / NCOL;
  int ri_x = kidx / 4800; int r2 = kidx % 4800;
  int q2   = r2 / 480;    int t  = (r2 % 480) / 32; int i = r2 & 31;
  int ri_o = col / 320;   int cq = col % 320;
  int q    = cq / 32;     int j  = cq & 31;
  int w_ri = ri_o ^ ri_x;
  const float* k4 = prm.k4 + (size_t)((((ri_o * 2 + ri_x) * 10 + q) * 10 + q2) * 16);
  float acc = 0.0f;
#pragma unroll
  for (int l1 = 0; l1 < 4; ++l1) {
    const float* wp = (w_ri == 0) ? prm.w[2 * l1] : prm.w[2 * l1 + 1];
#pragma unroll
    for (int m = 0; m <= l1; ++m) {
      float cf = k4[l1 * 4 + m];
      float wv = wp[(((size_t)(m * TT_ + t) * PP_ + p) * CIN_ + i) * COUT_ + j];
      acc = fmaf(cf, wv, acc);
    }
  }
  prm.weff[((size_t)p * NCOL + col) * KDIM + kidx] = (_Float16)acc;
}

// ---------------- Kernel 3: WMMA GEMM + bias epilogue ----------------
union HF { half16 h16; half8 h8[2]; };
union FV { fv8 f8; fv4 f4[2]; };

// A fragment (16x32 f16): lane<16 holds K {0..15}, lane>=16 holds {16..31}.
// r already includes the per-lane-half +16 element offset; loads 2x(two float4) + cvt.
__device__ inline half16 load_a_frag(const float* r) {
  FV lo, hi;
  lo.f4[0] = *(const fv4*)(r);
  lo.f4[1] = *(const fv4*)(r + 4);
  hi.f4[0] = *(const fv4*)(r + 8);
  hi.f4[1] = *(const fv4*)(r + 12);
  HF o;
  o.h8[0] = __builtin_convertvector(lo.f8, half8);
  o.h8[1] = __builtin_convertvector(hi.f8, half8);
  return o.h16;
}

// B fragment (32x16 f16): lane = column (lane%16); 16 contiguous K halves
// starting at kb*32 + (lane>=16 ? 16 : 0). Weff stored col-major => contiguous.
__device__ inline half16 load_b_frag(const _Float16* bp) {
  HF o;
  o.h8[0] = *(const half8*)(bp);
  o.h8[1] = *(const half8*)(bp + 8);
  return o.h16;
}

#define WMMA_F16(ACC, A, B) \
  (ACC) = __builtin_amdgcn_wmma_f32_16x16x32_f16(false, (A), false, (B), (short)0, (ACC), false, false)

__launch_bounds__(256)
__global__ void gemm_kernel(Params prm) {
  const int p    = blockIdx.z;         // 0..4
  const int nt   = blockIdx.y;         // 0..4   (N tiles of 128)
  const int mt   = blockIdx.x;         // 0..3   (M tiles of 256)
  const int tid  = threadIdx.x;
  const int wave = tid >> 5;           // 8 waves stacked along M
  const int lane = tid & 31;
  const int lr   = lane & 15;
  const int lh   = lane >> 4;

  const int mBase = mt * 256 + wave * 32;   // wave tile: 32 (M) x 128 (N)
  const int nBase = nt * 128;

  v8f acc[2][8] = {};

  // Clamp row indices (M padded 1000 -> 1024); stores are masked below.
  int n0 = mBase + lr;       if (n0 > NT_ - 1) n0 = NT_ - 1;
  int n1 = mBase + 16 + lr;  if (n1 > NT_ - 1) n1 = NT_ - 1;

  // B pointer: one base, all 8 fragments via immediate offsets (nf*16*KDIM halves),
  // advanced by 32 halves (64 B) per K-step.
  const _Float16* bp = prm.weff + (size_t)p * NCOL * KDIM
                     + (size_t)(nBase + lr) * KDIM + lh * 16;

  const Group* grp = prm.grp;

  for (int g = 0; g < NGROUPS; ++g) {
    Group gr = grp[g];
    // Hoist the 64-bit row-stride multiplies out of the hot loop.
    const float* r0 = gr.ptr + (size_t)n0 * gr.rowStride + p * CIN_ + lh * 16;
    const float* r1 = gr.ptr + (size_t)n1 * gr.rowStride + p * CIN_ + lh * 16;
    const size_t step = (size_t)gr.tStep;

    for (int t = 0; t < TT_; ++t) {
      half16 A0 = load_a_frag(r0);
      half16 A1 = load_a_frag(r1);

      // Prefetch next K-step of the B stream into near caches.
      __builtin_prefetch((const void*)(bp + 32), 0, 3);
      __builtin_prefetch((const void*)(bp + 4 * 16 * (size_t)KDIM + 32), 0, 3);

      // Batch 1: B0..B3 then 8 WMMAs (caps live B fragments at ~4).
      {
        half16 B0 = load_b_frag(bp);
        half16 B1 = load_b_frag(bp + 1 * 16 * (size_t)KDIM);
        half16 B2 = load_b_frag(bp + 2 * 16 * (size_t)KDIM);
        half16 B3 = load_b_frag(bp + 3 * 16 * (size_t)KDIM);
        WMMA_F16(acc[0][0], A0, B0); WMMA_F16(acc[1][0], A1, B0);
        WMMA_F16(acc[0][1], A0, B1); WMMA_F16(acc[1][1], A1, B1);
        WMMA_F16(acc[0][2], A0, B2); WMMA_F16(acc[1][2], A1, B2);
        WMMA_F16(acc[0][3], A0, B3); WMMA_F16(acc[1][3], A1, B3);
      }
      // Batch 2: B4..B7 then 8 WMMAs.
      {
        half16 B4 = load_b_frag(bp + 4 * 16 * (size_t)KDIM);
        half16 B5 = load_b_frag(bp + 5 * 16 * (size_t)KDIM);
        half16 B6 = load_b_frag(bp + 6 * 16 * (size_t)KDIM);
        half16 B7 = load_b_frag(bp + 7 * 16 * (size_t)KDIM);
        WMMA_F16(acc[0][4], A0, B4); WMMA_F16(acc[1][4], A1, B4);
        WMMA_F16(acc[0][5], A0, B5); WMMA_F16(acc[1][5], A1, B5);
        WMMA_F16(acc[0][6], A0, B6); WMMA_F16(acc[1][6], A1, B6);
        WMMA_F16(acc[0][7], A0, B7); WMMA_F16(acc[1][7], A1, B7);
      }

      r0 += step;
      r1 += step;
      bp += 32;
    }
  }

  // Epilogue: per N-fragment the column block is uniform -> scalar decode.
#pragma unroll
  for (int nf = 0; nf < 8; ++nf) {
    int col0 = nBase + nf * 16;
    int ri_o = col0 / 320;
    int cq   = col0 % 320;
    int q    = cq / 32;
    int j0   = cq & 31;       // 0 or 16
    int l    = q_to_deg(q);
    int lm   = q - l * (l + 1) / 2;
    float* op = prm.out + OUT_OFF[2 * l + ri_o];
    float bv = 0.0f;
    if (!(ri_o == 1 && lm == 0)) {   // imag bias masked at lm==0
      const float* bptr = prm.b[2 * l + ri_o];
      bv = bptr[(lm * PP_ + p) * COUT_ + j0 + lr];
    }
    const int lp1 = l + 1;
#pragma unroll
    for (int mf = 0; mf < 2; ++mf) {
#pragma unroll
      for (int r = 0; r < 8; ++r) {
        int n = mBase + mf * 16 + lh * 8 + r;   // C layout: VGPR r -> row r (+8 for hi lanes)
        if (n < NT_) {
          size_t off = (((size_t)n * lp1 + lm) * PP_ + p) * COUT_ + j0 + lr;
          op[off] = acc[mf][nf][r] + bv;
        }
      }
    }
  }
}

// ---------------- host launch ----------------
extern "C" void kernel_launch(void* const* d_in, const int* in_sizes, int n_in,
                              void* d_out, int out_size, void* d_ws, size_t ws_size,
                              hipStream_t stream) {
  (void)in_sizes; (void)n_in; (void)out_size;
  Params prm;
  for (int i = 0; i < 8; ++i) {
    prm.x[i] = (const float*)d_in[i];
    prm.w[i] = (const float*)d_in[8 + i];
    prm.b[i] = (const float*)d_in[16 + i];
  }
  prm.out = (float*)d_out;
  char* ws = (char*)d_ws;
  prm.k4   = (float*)ws;                       // 25.6 KB
  prm.grp  = (Group*)(ws + 32 * 1024);         // 480 B
  prm.weff = (_Float16*)(ws + 64 * 1024);      // 61.44 MB
  (void)ws_size;

  // 1) CG tables + K-group metadata (6400 + 20 threads)
  setup_kernel<<<27, 256, 0, stream>>>(prm);

  // 2) Effective weight matrix build: 5*640*9600 = 30.72M elements
  long long total = (long long)PP_ * NCOL * KDIM;
  int blocks = (int)((total + 255) / 256);
  build_weff<<<blocks, 256, 0, stream>>>(prm);

  // 3) WMMA GEMM: grid (Mtiles=4, Ntiles=5, p=5), 8 waves/block, wave tile 32x128
  dim3 g(4, 5, 5);
  gemm_kernel<<<g, 256, 0, stream>>>(prm);
}
